// DDiTBlock_7387343749470
// MI455X (gfx1250) — matrix operand order repaired and checked by reading, not compiled, not run on hardware
//
#include <hip/hip_runtime.h>
#include <math.h>
#include <stdint.h>

typedef __attribute__((ext_vector_type(16))) __bf16 v16bf;
typedef __attribute__((ext_vector_type(8)))  float  v8f;

#define DIMC   1024
#define SEQ    1024
#define BATCH  4
#define NHEAD  16
#define HDIM   64
#define MTOK   4096      // BATCH*SEQ
#define MODW   6144      // 6*DIM

// ---------------- workspace layout (bytes) ----------------
constexpr size_t OFF_WQKVT  = 0;                               // bf16 [3072][1024]
constexpr size_t OFF_WATTNT = OFF_WQKVT  + 3145728ull * 2;     // bf16 [1024][1024]
constexpr size_t OFF_WMLP1T = OFF_WATTNT + 1048576ull * 2;     // bf16 [4096][1024]
constexpr size_t OFF_WMLP2T = OFF_WMLP1T + 4194304ull * 2;     // bf16 [1024][4096]
constexpr size_t OFF_MOD    = OFF_WMLP2T + 4194304ull * 2;     // f32  [4][6144]
constexpr size_t OFF_H1     = OFF_MOD    + 24576ull   * 4;     // bf16 [4096][1024]
constexpr size_t OFF_QKV    = OFF_H1     + 4194304ull * 2;     // f32  [4096][3072]
constexpr size_t OFF_QB     = OFF_QKV    + 12582912ull* 4;     // bf16 [B,H,S,64]
constexpr size_t OFF_KB     = OFF_QB     + 4194304ull * 2;     // bf16 [B,H,S,64]
constexpr size_t OFF_VT     = OFF_KB     + 4194304ull * 2;     // bf16 [B,H,64,S]
constexpr size_t OFF_OB     = OFF_VT     + 4194304ull * 2;     // bf16 [4096][1024]
constexpr size_t OFF_X2     = OFF_OB     + 4194304ull * 2;     // f32  [4096][1024]
constexpr size_t OFF_H2     = OFF_X2     + 4194304ull * 4;     // bf16 [4096][1024]
constexpr size_t OFF_M1     = OFF_H2     + 4194304ull * 2;     // bf16 [4096][4096]

// ---------------- helpers ----------------
static __device__ inline unsigned short f2bf(float f) {
    unsigned int u = __float_as_uint(f);
    u += 0x7fffu + ((u >> 16) & 1u);          // round-to-nearest-even
    return (unsigned short)(u >> 16);
}

// A-fragment (16x32 bf16): lane = row (m = lane&15), hi = lane>>4.
// elements 0..7  = K  8*hi .. 8*hi+7 ; elements 8..15 = K 16+8*hi .. 16+8*hi+7
static __device__ inline v16bf loadA(const unsigned short* p, int hi) {
    union { uint4 u[2]; v16bf v; } r;
    r.u[0] = *reinterpret_cast<const uint4*>(p + 8 * hi);
    r.u[1] = *reinterpret_cast<const uint4*>(p + 16 + 8 * hi);
    return r.v;
}
// B-fragment (32x16 bf16) given B^T row (contiguous K of length 32):
// lane = col (n = lane&15); elements 0..7 = K 8*hi .. 8*hi+7 ; elements 8..15 = K 16+8*hi .. 16+8*hi+7
static __device__ inline v16bf loadB(const unsigned short* p, int hi) {
    union { uint4 u[2]; v16bf v; } r;
    r.u[0] = *reinterpret_cast<const uint4*>(p + 8 * hi);
    r.u[1] = *reinterpret_cast<const uint4*>(p + 16 + 8 * hi);
    return r.v;
}
static __device__ inline v8f wmma_bf16(v16bf a, v16bf b, v8f c) {
    return __builtin_amdgcn_wmma_f32_16x16x32_bf16(false, a, false, b, (short)0, c, false, false);
}
static __device__ inline float gelu_tanh(float x) {
    return 0.5f * x * (1.0f + tanhf(0.7978845608028654f * (x + 0.044715f * x * x * x)));
}

// ---------------- kernel: mod = c @ w_ada + b_ada ----------------
__global__ __launch_bounds__(256) void mod_kernel(const float* __restrict__ c,
                                                  const float* __restrict__ w_ada,
                                                  const float* __restrict__ b_ada,
                                                  float* __restrict__ mod) {
    int j = blockIdx.x * 256 + threadIdx.x;      // 0 .. 4*6144-1
    int b = j / MODW;
    int jj = j - b * MODW;
    float acc = b_ada[jj];
    const float* cr = c + (size_t)b * DIMC;
    for (int k = 0; k < DIMC; ++k)
        acc += cr[k] * w_ada[(size_t)k * MODW + jj];
    mod[j] = acc;
}

// ---------------- kernel: convert weight fp32 [K][N] -> bf16 transposed [N][K] ----------------
__global__ __launch_bounds__(256) void cvtT_kernel(const float* __restrict__ w,
                                                   unsigned short* __restrict__ out,
                                                   int K, int N) {
    size_t idx = (size_t)blockIdx.x * 256 + threadIdx.x;  // over K*N
    int n = (int)(idx / (size_t)K);
    int k = (int)(idx - (size_t)n * K);
    out[idx] = f2bf(w[(size_t)k * N + n]);
}

// ---------------- kernel: LayerNorm + adaLN modulation -> bf16 ----------------
__global__ __launch_bounds__(256) void ln_mod_kernel(const float* __restrict__ xin,
                                                     const float* __restrict__ wln,
                                                     const float* __restrict__ mod,
                                                     int shiftChunk, int scaleChunk,
                                                     unsigned short* __restrict__ out) {
    __shared__ float s1[256], s2[256];
    int tok = blockIdx.x, t = threadIdx.x;
    const float* row = xin + (size_t)tok * DIMC;
    float a0 = row[t], a1 = row[t + 256], a2 = row[t + 512], a3 = row[t + 768];
    s1[t] = a0 + a1 + a2 + a3;
    s2[t] = a0 * a0 + a1 * a1 + a2 * a2 + a3 * a3;
    __syncthreads();
    for (int o = 128; o > 0; o >>= 1) {
        if (t < o) { s1[t] += s1[t + o]; s2[t] += s2[t + o]; }
        __syncthreads();
    }
    float mu = s1[0] * (1.0f / DIMC);
    float var = s2[0] * (1.0f / DIMC) - mu * mu;
    float rstd = rsqrtf(var + 1e-5f);
    int b = tok >> 10;
    const float* shp = mod + (size_t)b * MODW + shiftChunk * DIMC;
    const float* scp = mod + (size_t)b * MODW + scaleChunk * DIMC;
    unsigned short* orow = out + (size_t)tok * DIMC;
    for (int i = t; i < DIMC; i += 256) {
        float v = (row[i] - mu) * rstd * wln[i];
        v = v * (1.0f + scp[i]) + shp[i];
        orow[i] = f2bf(v);
    }
}

// ---------------- kernel: RoPE split + layout for attention ----------------
// Q is pre-scaled by 1/sqrt(HDIM) so the attention loop needs no score scaling.
__global__ __launch_bounds__(256) void rope_kernel(const float* __restrict__ qkv,
                                                   const float* __restrict__ cosp,
                                                   const float* __restrict__ sinp,
                                                   unsigned short* __restrict__ qb,
                                                   unsigned short* __restrict__ kb,
                                                   unsigned short* __restrict__ vt) {
    int idx = blockIdx.x * 256 + threadIdx.x;    // B*S*H*32
    int i = idx & 31;
    int h = (idx >> 5) & 15;
    int s = (idx >> 9) & 1023;
    int b = idx >> 19;
    const float* rowq = qkv + (size_t)(b * SEQ + s) * (3 * DIMC);
    float cv = cosp[s * 32 + i], sv = sinp[s * 32 + i];
    int d1 = h * HDIM + i, d2 = d1 + 32;
    float q1 = rowq[d1],          q2 = rowq[d2];
    float k1 = rowq[DIMC + d1],   k2 = rowq[DIMC + d2];
    float v1 = rowq[2*DIMC + d1], v2 = rowq[2*DIMC + d2];
    size_t bh = (size_t)b * NHEAD + h;
    size_t qo = (bh * SEQ + s) * HDIM;
    const float qs = 0.125f;   // 1/sqrt(64)
    qb[qo + i]      = f2bf((q1 * cv - q2 * sv) * qs);
    qb[qo + 32 + i] = f2bf((q2 * cv + q1 * sv) * qs);
    kb[qo + i]      = f2bf(k1 * cv - k2 * sv);
    kb[qo + 32 + i] = f2bf(k2 * cv + k1 * sv);
    vt[(bh * HDIM + i) * SEQ + s]      = f2bf(v1);
    vt[(bh * HDIM + 32 + i) * SEQ + s] = f2bf(v2);
}

// ---------------- kernel: tiled bf16 WMMA GEMM, async-LDS double buffered ----------------
// EPI 0: store f32 C        EPI 2: bf16(gelu(C+bias))
// EPI 3: f32 resid + gate*C EPI 4: f32 resid + gate*(C+bias)
template <int EPI>
__global__ __launch_bounds__(256) void gemm_kernel(const unsigned short* __restrict__ A,
                                                   const unsigned short* __restrict__ Bt,
                                                   int Ktot, int Ntot,
                                                   float* __restrict__ outF,
                                                   unsigned short* __restrict__ outH,
                                                   const float* __restrict__ bias,
                                                   const float* __restrict__ resid,
                                                   const float* __restrict__ mod,
                                                   int modChunk) {
    __shared__ unsigned short sA[2][128 * 48];
    __shared__ unsigned short sB[2][128 * 48];
    int t = threadIdx.x;
    int w = t >> 5, lane = t & 31, ln = lane & 15, hi = lane >> 4;
    int rowBase = blockIdx.y * 128, colBase = blockIdx.x * 128;

    v8f acc[8];
    v8f z = {0.f, 0.f, 0.f, 0.f, 0.f, 0.f, 0.f, 0.f};
#pragma unroll
    for (int i = 0; i < 8; ++i) acc[i] = z;

    // per-thread staging slot: row rl, 16-element k-chunk kc (32B = 2 x b128)
    int rl = t >> 1, kc = (t & 1) * 16;
    const unsigned short* gA = A  + (size_t)(rowBase + rl) * Ktot + kc;
    const unsigned short* gB = Bt + (size_t)(colBase + rl) * Ktot + kc;
    unsigned laA = (unsigned)(uintptr_t)&sA[0][rl * 48 + kc];
    unsigned laB = (unsigned)(uintptr_t)&sB[0][rl * 48 + kc];
    const unsigned BUFO = 128u * 48u * 2u;     // bytes per buffer

    // async global->LDS: offset: applies to BOTH global and LDS addresses
#define ISSUE_TILE(bufsel, koff)                                               \
    {                                                                          \
        unsigned a0 = laA + (unsigned)(bufsel) * BUFO;                         \
        unsigned b0 = laB + (unsigned)(bufsel) * BUFO;                         \
        unsigned long long ga = (unsigned long long)(uintptr_t)(gA + (koff));  \
        unsigned long long gb = (unsigned long long)(uintptr_t)(gB + (koff));  \
        asm volatile(                                                          \
            "global_load_async_to_lds_b128 %0, %1, off\n\t"                    \
            "global_load_async_to_lds_b128 %0, %1, off offset:16\n\t"          \
            "global_load_async_to_lds_b128 %2, %3, off\n\t"                    \
            "global_load_async_to_lds_b128 %2, %3, off offset:16"              \
            :: "v"(a0), "v"(ga), "v"(b0), "v"(gb) : "memory");                 \
    }

    ISSUE_TILE(0, 0)
    for (int k0 = 0; k0 < Ktot; k0 += 32) {
        int cur = (k0 >> 5) & 1;
        if (k0 + 32 < Ktot) {
            ISSUE_TILE(cur ^ 1, k0 + 32)
            asm volatile("s_wait_asynccnt 0x4" ::: "memory");  // current tile landed
        } else {
            asm volatile("s_wait_asynccnt 0x0" ::: "memory");
        }
        __syncthreads();

        v16bf af = loadA(&sA[cur][(16 * w + ln) * 48], hi);
#pragma unroll
        for (int t8 = 0; t8 < 8; ++t8) {
            v16bf bf_ = loadB(&sB[cur][(16 * t8 + ln) * 48], hi);
            acc[t8] = wmma_bf16(af, bf_, acc[t8]);
        }
        __syncthreads();   // all waves done reading before this buffer is reloaded
    }
#undef ISSUE_TILE

#pragma unroll
    for (int t8 = 0; t8 < 8; ++t8) {
        int col = colBase + 16 * t8 + ln;
#pragma unroll
        for (int r = 0; r < 8; ++r) {
            int row = rowBase + 16 * w + 8 * hi + r;
            float v = acc[t8][r];
            size_t oi = (size_t)row * Ntot + col;
            if (EPI == 0) {
                outF[oi] = v;
            } else if (EPI == 2) {
                v = gelu_tanh(v + bias[col]);
                outH[oi] = f2bf(v);
            } else if (EPI == 3) {
                float g = mod[(size_t)(row >> 10) * MODW + modChunk * DIMC + col];
                outF[oi] = resid[oi] + g * v;
            } else if (EPI == 4) {
                v += bias[col];
                float g = mod[(size_t)(row >> 10) * MODW + modChunk * DIMC + col];
                outF[oi] = resid[oi] + g * v;
            }
        }
    }
}

// ---------------- kernel: flash attention, one wave per 16-query tile ----------------
__global__ __launch_bounds__(128) void attn_kernel(const unsigned short* __restrict__ q,
                                                   const unsigned short* __restrict__ k,
                                                   const unsigned short* __restrict__ vt,
                                                   unsigned short* __restrict__ o) {
    __shared__ unsigned short sP[4][16 * 48];     // per-wave P transpose scratch
    int lane = threadIdx.x & 31, w = threadIdx.x >> 5;
    int ln = lane & 15, hi = lane >> 4;
    int tile = blockIdx.x * 4 + w;                // B*H*(S/16) tiles
    int bh = tile >> 6;                           // S/16 = 64 tiles per (b,h)
    int qbase = (tile & 63) << 4;

    const unsigned short* qp = q + ((size_t)bh * SEQ + qbase + ln) * HDIM;
    v16bf qa0 = loadA(qp, hi);        // d 0..31 (pre-scaled by 1/8)
    v16bf qa1 = loadA(qp + 32, hi);   // d 32..63

    v8f z = {0.f, 0.f, 0.f, 0.f, 0.f, 0.f, 0.f, 0.f};
    v8f oacc[4];
#pragma unroll
    for (int i = 0; i < 4; ++i) oacc[i] = z;
    float mi[8], li[8];
#pragma unroll
    for (int r = 0; r < 8; ++r) { mi[r] = -1e30f; li[r] = 0.f; }

    const unsigned short* kbase = k  + (size_t)bh * SEQ * HDIM;
    const unsigned short* vbase = vt + (size_t)bh * HDIM * SEQ;
    unsigned short* myP = &sP[w][0];

    for (int n0 = 0; n0 < SEQ; n0 += 32) {
        // scores: 16 rows x 32 keys, D=64 split into two k-halves
        v8f s0 = z, s1 = z;
        const unsigned short* kp0 = kbase + (size_t)(n0 + ln) * HDIM;
        s0 = wmma_bf16(qa0, loadB(kp0, hi), s0);
        s0 = wmma_bf16(qa1, loadB(kp0 + 32, hi), s0);
        const unsigned short* kp1 = kbase + (size_t)(n0 + 16 + ln) * HDIM;
        s1 = wmma_bf16(qa0, loadB(kp1, hi), s1);
        s1 = wmma_bf16(qa1, loadB(kp1 + 32, hi), s1);

        // online softmax per row (rows m = 8*hi + r live in vgpr slot r)
#pragma unroll
        for (int r = 0; r < 8; ++r) {
            float a = s0[r];
            float b = s1[r];
            float pm = fmaxf(a, b);
            for (int off = 1; off < 16; off <<= 1)
                pm = fmaxf(pm, __shfl_xor(pm, off, 32));
            float mn = fmaxf(mi[r], pm);
            float alpha = __expf(mi[r] - mn);
            float e0 = __expf(a - mn);
            float e1 = __expf(b - mn);
            float rs = e0 + e1;
            for (int off = 1; off < 16; off <<= 1)
                rs += __shfl_xor(rs, off, 32);
            li[r] = li[r] * alpha + rs;
            mi[r] = mn;
#pragma unroll
            for (int dt = 0; dt < 4; ++dt) oacc[dt][r] *= alpha;
            int prow = 8 * hi + r;
            myP[prow * 48 + ln]      = f2bf(e0);
            myP[prow * 48 + 16 + ln] = f2bf(e1);
        }
        asm volatile("s_wait_dscnt 0" ::: "memory");   // wave-local LDS RAW fence

        // O += P (16x32) @ V (32x64): V^T rows give contiguous-K B fragments
        v16bf pa = loadA(myP + ln * 48, hi);
#pragma unroll
        for (int dt = 0; dt < 4; ++dt) {
            const unsigned short* vp = vbase + (size_t)(dt * 16 + ln) * SEQ + n0;
            oacc[dt] = wmma_bf16(pa, loadB(vp, hi), oacc[dt]);
        }
        asm volatile("s_wait_dscnt 0" ::: "memory");   // reads done before next-step stores
    }

    int b = bh >> 4, h = bh & 15;
#pragma unroll
    for (int dt = 0; dt < 4; ++dt) {
#pragma unroll
        for (int r = 0; r < 8; ++r) {
            float val = oacc[dt][r] / li[r];
            size_t row = (size_t)b * SEQ + qbase + 8 * hi + r;
            o[row * DIMC + h * HDIM + dt * 16 + ln] = f2bf(val);
        }
    }
}

// ---------------- host launcher ----------------
extern "C" void kernel_launch(void* const* d_in, const int* in_sizes, int n_in,
                              void* d_out, int out_size, void* d_ws, size_t ws_size,
                              hipStream_t stream) {
    const float* x        = (const float*)d_in[0];
    const float* cosp     = (const float*)d_in[1];
    const float* sinp     = (const float*)d_in[2];
    const float* c        = (const float*)d_in[3];
    const float* w_ln1    = (const float*)d_in[4];
    const float* w_ln2    = (const float*)d_in[5];
    const float* w_qkv    = (const float*)d_in[6];
    const float* w_attn   = (const float*)d_in[7];
    const float* w_mlp1   = (const float*)d_in[8];
    const float* b_mlp1   = (const float*)d_in[9];
    const float* w_mlp2   = (const float*)d_in[10];
    const float* b_mlp2   = (const float*)d_in[11];
    const float* w_ada    = (const float*)d_in[12];
    const float* b_ada    = (const float*)d_in[13];

    char* ws = (char*)d_ws;
    unsigned short* WQKVT  = (unsigned short*)(ws + OFF_WQKVT);
    unsigned short* WATTNT = (unsigned short*)(ws + OFF_WATTNT);
    unsigned short* WMLP1T = (unsigned short*)(ws + OFF_WMLP1T);
    unsigned short* WMLP2T = (unsigned short*)(ws + OFF_WMLP2T);
    float*          MODP   = (float*)(ws + OFF_MOD);
    unsigned short* H1     = (unsigned short*)(ws + OFF_H1);
    float*          QKV    = (float*)(ws + OFF_QKV);
    unsigned short* QB     = (unsigned short*)(ws + OFF_QB);
    unsigned short* KB     = (unsigned short*)(ws + OFF_KB);
    unsigned short* VT     = (unsigned short*)(ws + OFF_VT);
    unsigned short* OB     = (unsigned short*)(ws + OFF_OB);
    float*          X2     = (float*)(ws + OFF_X2);
    unsigned short* H2     = (unsigned short*)(ws + OFF_H2);
    unsigned short* M1     = (unsigned short*)(ws + OFF_M1);
    float*          OUT    = (float*)d_out;

    // weight conversions (bf16, transposed [N][K])
    cvtT_kernel<<<(1024 * 3072) / 256, 256, 0, stream>>>(w_qkv,  WQKVT,  1024, 3072);
    cvtT_kernel<<<(1024 * 1024) / 256, 256, 0, stream>>>(w_attn, WATTNT, 1024, 1024);
    cvtT_kernel<<<(1024 * 4096) / 256, 256, 0, stream>>>(w_mlp1, WMLP1T, 1024, 4096);
    cvtT_kernel<<<(4096 * 1024) / 256, 256, 0, stream>>>(w_mlp2, WMLP2T, 4096, 1024);

    // adaLN modulation vector
    mod_kernel<<<(BATCH * MODW) / 256, 256, 0, stream>>>(c, w_ada, b_ada, MODP);

    // LN1 + msa modulation -> bf16
    ln_mod_kernel<<<MTOK, 256, 0, stream>>>(x, w_ln1, MODP, 0, 1, H1);

    // QKV projection (f32 out for RoPE precision)
    gemm_kernel<0><<<dim3(3072 / 128, MTOK / 128), 256, 0, stream>>>(
        H1, WQKVT, 1024, 3072, QKV, nullptr, nullptr, nullptr, nullptr, 0);

    // RoPE + attention layouts (Q/K row-major per head, V transposed)
    rope_kernel<<<(BATCH * SEQ * NHEAD * 32) / 256, 256, 0, stream>>>(
        QKV, cosp, sinp, QB, KB, VT);

    // flash attention: 4096 query tiles, 4 waves per block
    attn_kernel<<<(BATCH * NHEAD * (SEQ / 16)) / 4, 128, 0, stream>>>(QB, KB, VT, OB);

    // output projection + gate_msa residual -> X2 (f32)
    gemm_kernel<3><<<dim3(1024 / 128, MTOK / 128), 256, 0, stream>>>(
        OB, WATTNT, 1024, 1024, X2, nullptr, nullptr, x, MODP, 2);

    // LN2 + mlp modulation -> bf16
    ln_mod_kernel<<<MTOK, 256, 0, stream>>>(X2, w_ln2, MODP, 3, 4, H2);

    // MLP up + GELU -> bf16
    gemm_kernel<2><<<dim3(4096 / 128, MTOK / 128), 256, 0, stream>>>(
        H2, WMLP1T, 1024, 4096, nullptr, M1, b_mlp1, nullptr, nullptr, 0);

    // MLP down + bias + gate_mlp residual -> final output
    gemm_kernel<4><<<dim3(1024 / 128, MTOK / 128), 256, 0, stream>>>(
        M1, WMLP2T, 4096, 1024, OUT, nullptr, b_mlp2, X2, MODP, 5);
}
